// LFA_72464688218274
// MI455X (gfx1250) — matrix operand order repaired and checked by reading, not compiled, not run on hardware
//
#include <hip/hip_runtime.h>

typedef _Float16 half_t;
typedef __attribute__((ext_vector_type(16))) _Float16 v16h;
typedef __attribute__((ext_vector_type(8)))  float    v8f;
typedef __attribute__((ext_vector_type(4)))  unsigned int u32x4;

#define NPTS   40960
#define KNB    16
#define NBATCH 2
#define BN     (NBATCH * NPTS)
#define NCONV  20

// conv order (jax tree-flatten, dict keys sorted):
// b1: fa.mlp, fa.mlp1, fa.mlp2, fa.mlp3[0], fa.mlp3[1], fa.mlp4[0], fa.mlp4[1], mlp1, mlp2, shortcut
// b2: same +10.  Each conv contributes 6 leaves: W, b, be, g, m, v.
constexpr int C_OUT[NCONV] = {32,16,16,16,16,16,16,16,32,32, 64,32,32,32,32,32,32,32,64,64};
constexpr int C_IN [NCONV] = {32,10,48,16,16,16,16, 8,16, 8, 64,10,96,32,32,32,32,32,32,32};
constexpr int pad16i(int x){ return (x+15)/16*16; }
constexpr int pad32i(int x){ return (x+31)/32*32; }
constexpr int wOff(int ci){ int o=0; for(int i=0;i<ci;++i) o += pad16i(C_OUT[i])*pad32i(C_IN[i]); return o; }
constexpr int bOff(int ci){ int o=0; for(int i=0;i<ci;++i) o += pad16i(C_OUT[i]); return o; }
constexpr int W_TOTAL = wOff(NCONV);   // 24576 f16 elems
constexpr int B_TOTAL = bOff(NCONV);   // 624 f32 elems

struct ParamPtrs { const float* p[NCONV*6]; };

#define DSWAIT() asm volatile("s_wait_dscnt 0x0" ::: "memory")

// ---------------- WMMA fragment helpers (CDNA5 wave32 layouts) ----------------
__device__ __forceinline__ v8f wmma16(v16h a, v16h b, v8f c) {
  return __builtin_amdgcn_wmma_f32_16x16x32_f16(false, a, false, b, (short)0, c, false, false);
}
// A 16x32 f16, M=lane&15 (+mtile*16); lanes<16: K {0..7}U{16..23}, lanes>=16: +8
__device__ __forceinline__ v16h load_a(const half_t* w, int cinp, int mtile, int kc, int lane){
  const half_t* base = w + (size_t)(mtile*16 + (lane & 15))*cinp + kc*32 + ((lane>>4)<<3);
  union { v16h h; u32x4 u[2]; } r;
  r.u[0] = *(const u32x4*)(base);
  r.u[1] = *(const u32x4*)(base + 16);
  return r.h;
}
// B 32x16 f16, N=lane&15; lanes<16: K {0..7}U{16..23}, lanes>=16: +8
__device__ __forceinline__ v16h load_b(const half_t* X, int cbase, int kc, int lane){
  const half_t* base = X + (lane & 15)*128 + cbase + kc*32 + ((lane>>4)<<3);
  union { v16h h; u32x4 u[2]; } r;
  r.u[0] = *(const u32x4*)(base);
  r.u[1] = *(const u32x4*)(base + 16);
  return r.h;
}
// C/D f32 16x16: lane n=lane&15, vgpr r -> M = mtile*16 + r + (lane>=16 ? 8 : 0)
__device__ __forceinline__ v8f load_bias8(const float* b, int mtile, int lane){
  const float* p = b + mtile*16 + ((lane>>4)<<3);
  v8f c;
#pragma unroll
  for (int r=0;r<8;++r) c[r] = p[r];
  return c;
}
__device__ __forceinline__ void store_cf16(half_t* X, int cbase, int mtile, int lane, v8f c){
  half_t* p = X + (lane & 15)*128 + cbase + mtile*16 + ((lane>>4)<<3);
  union { u32x4 u; half_t h[8]; } t;
#pragma unroll
  for (int r=0;r<8;++r) t.h[r] = (half_t)c[r];
  *(u32x4*)p = t.u;
}
__device__ __forceinline__ v8f relu8(v8f c){
#pragma unroll
  for (int r=0;r<8;++r) c[r] = fmaxf(c[r], 0.f);
  return c;
}

// ---------------- prep: fold BN into conv, pack f16 padded weights ----------------
template<int CI>
__global__ void prep_conv(ParamPtrs pp, half_t* __restrict__ wf, float* __restrict__ bf) {
  constexpr int CO  = C_OUT[CI], CIn = C_IN[CI];
  constexpr int COP = pad16i(CO), CIP = pad32i(CIn);
  const float* W  = pp.p[CI*6+0];
  const float* b  = pp.p[CI*6+1];
  const float* be = pp.p[CI*6+2];
  const float* g  = pp.p[CI*6+3];
  const float* m  = pp.p[CI*6+4];
  const float* v  = pp.p[CI*6+5];
  half_t* wo = wf + wOff(CI);
  float*  bo = bf + bOff(CI);
  for (int o = threadIdx.x; o < COP; o += blockDim.x) {
    float s = 0.f, bb = 0.f;
    if (o < CO) { s = g[o] * rsqrtf(v[o] + 1e-5f); bb = (b[o] - m[o]) * s + be[o]; }
    bo[o] = bb;
    for (int c = 0; c < CIP; ++c) {
      float wv = (o < CO && c < CIn) ? W[o*CIn + c] * s : 0.f;
      wo[(size_t)o*CIP + c] = (half_t)wv;
    }
  }
}
template<int CI>
static void launch_prep(ParamPtrs pp, half_t* wf, float* bf, hipStream_t s) {
  prep_conv<CI><<<1, 64, 0, s>>>(pp, wf, bf);
  if constexpr (CI + 1 < NCONV) launch_prep<CI+1>(pp, wf, bf, s);
}

// ---------------- pointwise CBR (block mlp1): [B][CIN][N] f32 -> [B*N][COUT] f16 ----------------
template<int CI, int CINc, int COUTc>
__global__ __launch_bounds__(256) void pointwise_cbr(
    const float* __restrict__ in, half_t* __restrict__ out,
    const half_t* __restrict__ wgl, const float* __restrict__ bgl)
{
  constexpr int CIP = pad32i(CINc);
  __shared__ __align__(16) half_t w[COUTc*CIP];
  __shared__ float bb[COUTc];
  for (int i = threadIdx.x; i < COUTc*CIP; i += 256) w[i] = wgl[wOff(CI) + i];
  for (int i = threadIdx.x; i < COUTc;     i += 256) bb[i] = bgl[bOff(CI) + i];
  __syncthreads();
  int p = blockIdx.x * 256 + threadIdx.x;
  int b = p / NPTS, n = p - b * NPTS;
  float xv[CINc];
#pragma unroll
  for (int c = 0; c < CINc; ++c) xv[c] = in[((size_t)b*CINc + c)*NPTS + n];
  __align__(16) half_t ov[COUTc];
#pragma unroll
  for (int o = 0; o < COUTc; ++o) {
    float a = bb[o];
#pragma unroll
    for (int c = 0; c < CINc; ++c) a += (float)w[o*CIP + c] * xv[c];
    ov[o] = (half_t)fmaxf(a, 0.f);
  }
#pragma unroll
  for (int o = 0; o < COUTc; o += 8)
    *(u32x4*)(out + (size_t)p*COUTc + o) = *(const u32x4*)(ov + o);
}

// ---------------- fused LFA block: one wave32 per point, PITER points per wave ----------------
// LDS activation tile per wave: 16 neighbors x 128 channels (f16).
// ch 0..9 geometry (pad->32, re-zeroed each iter) | s2 out f_xyz at 64+D | gather f_nb at 64 |
// mlp out at 0..2D-1 | f_tile at 2D | f at 96
template<int D, int DINc, int DOUTc, int CB>
__global__ __launch_bounds__(128) void lfa_block(
    const float* __restrict__ xyz, const int* __restrict__ nidx,
    const half_t* __restrict__ f1, const float* __restrict__ xin,
    float* __restrict__ xout,
    const half_t* __restrict__ wgl, const float* __restrict__ bgl)
{
  constexpr int WAVES = 4;
  constexpr int PITER = 8;          // points per wave: amortize weight-copy + launch
  constexpr int WEL = wOff(CB+10) - wOff(CB);
  constexpr int BEL = bOff(CB+10) - bOff(CB);
  __shared__ __align__(16) half_t lw[WEL];
  __shared__ __align__(16) float  lb[BEL];
  __shared__ __align__(16) half_t lx[WAVES][16][128];
  __shared__ __align__(16) float  lp[WAVES][64];

  for (int i = threadIdx.x; i < WEL; i += 128) lw[i] = wgl[wOff(CB) + i];
  for (int i = threadIdx.x; i < BEL; i += 128) lb[i] = bgl[bOff(CB) + i];
  __syncthreads();

  const int lane = threadIdx.x & 31;
  const int wave = threadIdx.x >> 5;
  const int gp0 = (blockIdx.x * WAVES + wave) * PITER;
  half_t* X = &lx[wave][0][0];
  float*  P = &lp[wave][0];

  // one-time zero of activation scratch (covers pads never written in-loop)
  {
    u32x4 z = (u32x4){0u,0u,0u,0u};
    u32x4* xp = (u32x4*)X;
#pragma unroll
    for (int i = 0; i < 8; ++i) xp[lane + 32*i] = z;
  }

  for (int it = 0; it < PITER; ++it) {
    const int gp = gp0 + it;
    const int b = gp / NPTS;
    const int n = gp - b * NPTS;

    // prefetch next iteration's gather inputs (overlaps this iter's WMMA chain)
    {
      const int gpn = (gp + 1 < BN) ? gp + 1 : gp;
      __builtin_prefetch(nidx + (size_t)gpn * KNB, 0, 0);
      __builtin_prefetch(f1 + (size_t)gpn * D, 0, 0);
    }

    // geometry features (ch 0..9; ch 10..31 re-zeroed) + neighbor-feature gather
    if (lane < 16) {
      const int idx = nidx[(size_t)(b*NPTS + n)*KNB + lane];
      const float* pt = xyz + (size_t)(b*NPTS + n)*3;
      const float* nb = xyz + (size_t)(b*NPTS + idx)*3;
      float tx=pt[0], ty=pt[1], tz=pt[2];
      float nx=nb[0], ny=nb[1], nz=nb[2];
      float rx=tx-nx, ry=ty-ny, rz=tz-nz;
      float dd = sqrtf(rx*rx + ry*ry + rz*rz);
      half_t* row = X + lane*128;
      union { u32x4 u; half_t h[8]; } g0, g1;
      g0.h[0]=(half_t)dd; g0.h[1]=(half_t)rx; g0.h[2]=(half_t)ry; g0.h[3]=(half_t)rz;
      g0.h[4]=(half_t)tx; g0.h[5]=(half_t)ty; g0.h[6]=(half_t)tz; g0.h[7]=(half_t)nx;
      *(u32x4*)row = g0.u;
      g1.h[0]=(half_t)ny; g1.h[1]=(half_t)nz;
#pragma unroll
      for (int r=2;r<8;++r) g1.h[r]=(half_t)0.f;
      *(u32x4*)(row + 8) = g1.u;
      u32x4 z = (u32x4){0u,0u,0u,0u};
      *(u32x4*)(row + 16) = z;   // re-zero pad ch 16..23
      *(u32x4*)(row + 24) = z;   // re-zero pad ch 24..31
      const half_t* fnb = f1 + (size_t)(b*NPTS + idx)*D;
#pragma unroll
      for (int c = 0; c < D; c += 8)
        *(u32x4*)(row + 64 + c) = *(const u32x4*)(fnb + c);
    }
    DSWAIT();

    // fa.mlp1: 10(pad32) -> D, out at ch 64+D
    {
      const half_t* w = lw + (wOff(CB+1) - wOff(CB));
      const float*  bv = lb + (bOff(CB+1) - bOff(CB));
      v16h bfr = load_b(X, 0, 0, lane);
#pragma unroll
      for (int mt = 0; mt < D/16; ++mt) {
        v8f c = load_bias8(bv, mt, lane);
        c = wmma16(load_a(w, 32, mt, 0, lane), bfr, c);
        store_cf16(X, 64 + D, mt, lane, relu8(c));
      }
    }
    DSWAIT();

    // fa.mlp: 2D -> 2D, in at ch 64, out at ch 0
    {
      constexpr int CINP = 2*D;
      constexpr int KC = CINP/32;
      const half_t* w = lw + (wOff(CB+0) - wOff(CB));
      const float*  bv = lb + (bOff(CB+0) - bOff(CB));
      v16h bfr[KC];
#pragma unroll
      for (int kc = 0; kc < KC; ++kc) bfr[kc] = load_b(X, 64, kc, lane);
#pragma unroll
      for (int mt = 0; mt < (2*D)/16; ++mt) {
        v8f c = load_bias8(bv, mt, lane);
#pragma unroll
        for (int kc = 0; kc < KC; ++kc) c = wmma16(load_a(w, CINP, mt, kc, lane), bfr[kc], c);
        store_cf16(X, 0, mt, lane, relu8(c));
      }
    }
    DSWAIT();

    // f_tile broadcast at ch 2D
    if (lane < 16) {
      const half_t* ftl = f1 + (size_t)(b*NPTS + n)*D;
      half_t* row = X + lane*128;
#pragma unroll
      for (int c = 0; c < D; c += 8)
        *(u32x4*)(row + 2*D + c) = *(const u32x4*)(ftl + c);
    }
    DSWAIT();

    // fa.mlp2: 3D(pad) -> D ; keep f32 accum, mirror f16 at ch 96
    v8f acc[D/16];
    {
      constexpr int CINP = pad32i(3*D);
      constexpr int KC = CINP/32;
      const half_t* w = lw + (wOff(CB+2) - wOff(CB));
      const float*  bv = lb + (bOff(CB+2) - bOff(CB));
      v16h bfr[KC];
#pragma unroll
      for (int kc = 0; kc < KC; ++kc) bfr[kc] = load_b(X, 0, kc, lane);
#pragma unroll
      for (int mt = 0; mt < D/16; ++mt) {
        v8f c = load_bias8(bv, mt, lane);
#pragma unroll
        for (int kc = 0; kc < KC; ++kc) c = wmma16(load_a(w, CINP, mt, kc, lane), bfr[kc], c);
        c = relu8(c);
        acc[mt] = c;
        store_cf16(X, 96, mt, lane, c);
      }
    }
    DSWAIT();

    // fa.mlp3 residuals x2: f = f + relu(conv(f))
    {
      const half_t* w3[2] = { lw + (wOff(CB+3)-wOff(CB)), lw + (wOff(CB+4)-wOff(CB)) };
      const float*  b3[2] = { lb + (bOff(CB+3)-bOff(CB)), lb + (bOff(CB+4)-bOff(CB)) };
#pragma unroll
      for (int rr = 0; rr < 2; ++rr) {
        v16h bfr = load_b(X, 96, 0, lane);
#pragma unroll
        for (int mt = 0; mt < D/16; ++mt) {
          v8f c = load_bias8(b3[rr], mt, lane);
          c = wmma16(load_a(w3[rr], pad32i(D), mt, 0, lane), bfr, c);
          acc[mt] += relu8(c);
        }
#pragma unroll
        for (int mt = 0; mt < D/16; ++mt) store_cf16(X, 96, mt, lane, acc[mt]);
        DSWAIT();
      }
    }

    // max-pool over 16 neighbors (xor reduce inside each 16-lane half)
#pragma unroll
    for (int mt = 0; mt < D/16; ++mt) {
#pragma unroll
      for (int mask = 8; mask >= 1; mask >>= 1) {
#pragma unroll
        for (int r = 0; r < 8; ++r) {
          float o = __shfl_xor(acc[mt][r], mask, 32);
          acc[mt][r] = fmaxf(acc[mt][r], o);
        }
      }
    }
    if ((lane & 15) == 0) {
      const int hb = (lane >> 4) << 3;
#pragma unroll
      for (int mt = 0; mt < D/16; ++mt)
#pragma unroll
        for (int r = 0; r < 8; ++r)
          P[mt*16 + hb + r] = acc[mt][r];
    }
    DSWAIT();

    // fa.mlp4 residuals x2 on pooled vector (VALU)
    {
      const half_t* w4[2] = { lw + (wOff(CB+5)-wOff(CB)), lw + (wOff(CB+6)-wOff(CB)) };
      const float*  b4[2] = { lb + (bOff(CB+5)-bOff(CB)), lb + (bOff(CB+6)-bOff(CB)) };
#pragma unroll
      for (int rr = 0; rr < 2; ++rr) {
        float nv = 0.f;
        if (lane < D) {
          float a = b4[rr][lane];
#pragma unroll
          for (int c = 0; c < D; ++c) a += (float)w4[rr][lane*pad32i(D) + c] * P[c];
          nv = P[lane] + fmaxf(a, 0.f);
        }
        DSWAIT();
        if (lane < D) P[lane] = nv;
        DSWAIT();
      }
    }

    // block mlp2 (D->DOUT) + shortcut (DIN->DOUT), write [B][DOUT][N]
    {
      const half_t* w2 = lw + (wOff(CB+8) - wOff(CB));
      const float*  b2 = lb + (bOff(CB+8) - bOff(CB));
      const half_t* wsc = lw + (wOff(CB+9) - wOff(CB));
      const float*  bsc = lb + (bOff(CB+9) - bOff(CB));
      float xv[DINc];
#pragma unroll
      for (int c = 0; c < DINc; ++c) xv[c] = xin[((size_t)b*DINc + c)*NPTS + n];
      float pv[D];
#pragma unroll
      for (int c = 0; c < D; ++c) pv[c] = P[c];
#pragma unroll
      for (int i = 0; i < DOUTc/32; ++i) {
        const int o = lane + i*32;
        float a = b2[o];
#pragma unroll
        for (int c = 0; c < D; ++c) a += (float)w2[o*pad32i(D) + c] * pv[c];
        float s = bsc[o];
#pragma unroll
        for (int c = 0; c < DINc; ++c) s += (float)wsc[o*pad32i(DINc) + c] * xv[c];
        xout[((size_t)b*DOUTc + o)*NPTS + n] = fmaxf(a, 0.f) + fmaxf(s, 0.f);
      }
    }
  }
}

// ---------------- host ----------------
extern "C" void kernel_launch(void* const* d_in, const int* in_sizes, int n_in,
                              void* d_out, int out_size, void* d_ws, size_t ws_size,
                              hipStream_t stream)
{
  (void)in_sizes; (void)n_in; (void)out_size; (void)ws_size;
  const float* feature = (const float*)d_in[0];
  const float* xyz     = (const float*)d_in[1];
  const int*   nidx    = (const int*)d_in[2];
  ParamPtrs pp;
  for (int i = 0; i < NCONV*6; ++i) pp.p[i] = (const float*)d_in[3 + i];

  char* ws = (char*)d_ws;
  size_t off = 0;
  auto take = [&](size_t bytes) { char* r = ws + off; off = (off + bytes + 255) & ~(size_t)255; return r; };
  half_t* wf   = (half_t*)take((size_t)W_TOTAL * 2);
  float*  bfv  = (float*) take((size_t)B_TOTAL * 4);
  half_t* f1a  = (half_t*)take((size_t)BN * 16 * 2);   // block1 mlp1 output (f16)
  float*  out1 = (float*) take((size_t)BN * 32 * 4);   // block1 output [B][32][N]
  half_t* f1b  = (half_t*)take((size_t)BN * 32 * 2);   // block2 mlp1 output (f16)

  launch_prep<0>(pp, wf, bfv, stream);

  constexpr int PTS_PER_WG = 4 * 8;  // WAVES * PITER
  pointwise_cbr<7, 8, 16><<<BN/256, 256, 0, stream>>>(feature, f1a, wf, bfv);
  lfa_block<16, 8, 32, 0><<<BN/PTS_PER_WG, 128, 0, stream>>>(xyz, nidx, f1a, feature, out1, wf, bfv);
  pointwise_cbr<17, 32, 32><<<BN/256, 256, 0, stream>>>(out1, f1b, wf, bfv);
  lfa_block<32, 32, 64, 10><<<BN/PTS_PER_WG, 128, 0, stream>>>(xyz, nidx, f1b, out1, (float*)d_out, wf, bfv);
}
